// Attention_72103910965317
// MI455X (gfx1250) — matrix operand order repaired and checked by reading, not compiled, not run on hardware
//
#include <hip/hip_runtime.h>
#include <hip/hip_bf16.h>
#include <math.h>

// ---------------------------------------------------------------------------
// Spatial self-attention for MI455X (gfx1250), wave32 + WMMA f16->f32.
//   Stage 0: x -> f16 copy; w_qkv, w_out -> transposed f16 ([N][K]) copies.
//   Stage 1: qkv = xh @ wqkvT (16x64 tile/wave, 16 WMMA, K fully unrolled)
//   Stage 2: flash attention, 8 waves/block sharing K/V tiles via
//            double-buffered LDS; tiles staged with ASYNC global->LDS copies
//            (ASYNCcnt-tracked), fallback to b128+ds_store if unavailable.
//   Stage 3: out = att @ woutT + b_out (16x64 tile/wave, 16 WMMA)
// ---------------------------------------------------------------------------

typedef __attribute__((ext_vector_type(16))) _Float16 v16h;
typedef __attribute__((ext_vector_type(8)))  _Float16 v8h;
typedef __attribute__((ext_vector_type(8)))  float    v8f;

#define HEADS 4
#define DHEAD 32
#define NTOK  4096           // 64*64 tokens per batch image
#define BATCH 4
#define CCH   128            // channels
#define MROWS (BATCH * NTOK) // 16384 flattened rows
#define F_QKV 384            // 3*H*d
#define JTILE 64             // attention column tile staged in LDS
#define QROWS_BLK 128        // Q rows per block (8 waves x 16)

#if __has_builtin(__builtin_amdgcn_global_load_async_to_lds_b128) && \
    __has_builtin(__builtin_amdgcn_s_wait_asynccnt)
#define USE_ASYNC_LDS 1
typedef __attribute__((ext_vector_type(4))) int v4i;
typedef v4i __attribute__((address_space(1))) gv4i;  // global (device) int4
typedef v4i __attribute__((address_space(3))) lv4i;  // LDS (shared) int4
#else
#define USE_ASYNC_LDS 0
#endif

union V16U { v16h v; v8h h[2]; };

// Build a v16h A/B fragment from two aligned 16-byte chunks.
static __device__ __forceinline__ v16h load2x8(const _Float16* p0, const _Float16* p1) {
  V16U u;
  u.h[0] = *reinterpret_cast<const v8h*>(p0);
  u.h[1] = *reinterpret_cast<const v8h*>(p1);
  return u.v;
}

static __device__ __forceinline__ v8f wmma_f16(v16h a, v16h b, v8f c) {
  // (neg_a, A, neg_b, B, c_mod, C, reuse_a, reuse_b)
  return __builtin_amdgcn_wmma_f32_16x16x32_f16(false, a, false, b, (short)0, c,
                                                false, false);
}

// ---------------------------------------------------------------------------
// Stage 0a: straight f32 -> f16 convert (for x).
// ---------------------------------------------------------------------------
__global__ void cvt_f16_kernel(const float* __restrict__ src,
                               _Float16* __restrict__ dst, int n) {
  int i = blockIdx.x * blockDim.x + threadIdx.x;
  if (i < n) dst[i] = (_Float16)src[i];
}

// ---------------------------------------------------------------------------
// Stage 0b: transpose + convert: src[rows][cols] f32 -> dst[cols][rows] f16.
// Gives column-major weights so B-fragment loads are contiguous per lane.
// ---------------------------------------------------------------------------
__global__ void tcvt_f16_kernel(const float* __restrict__ src,
                                _Float16* __restrict__ dst, int rows, int cols) {
  int i = blockIdx.x * blockDim.x + threadIdx.x;
  if (i < rows * cols) {
    int r = i / cols, c = i % cols;
    dst[c * rows + r] = (_Float16)src[i];
  }
}

// ---------------------------------------------------------------------------
// Stage 1: qkv = xh @ w_qkv, 16(M) x 64(N) tile per wave, K=128 unrolled.
// wqkvT is [384][128] f16 (column-major weight). Writes q (pre-scaled) and k
// as [b,H,N,d], v transposed as [b,H,d,N].
// ---------------------------------------------------------------------------
__global__ void qkv_proj_kernel(const _Float16* __restrict__ xh,
                                const _Float16* __restrict__ wqkvT,
                                _Float16* __restrict__ q,
                                _Float16* __restrict__ kmat,
                                _Float16* __restrict__ vT) {
  const int lane = threadIdx.x & 31;
  const int g  = lane >> 4;   // lane group (0/1)
  const int ln = lane & 15;   // lane within group
  const int m0 = blockIdx.x * 16;   // flattened row (b*NTOK + token)
  const int n0 = blockIdx.y * 64;   // output feature block

  v8f acc[4] = {};
  const _Float16* xr = xh + (size_t)(m0 + ln) * CCH;  // A row for this lane
#pragma unroll
  for (int k0 = 0; k0 < CCH; k0 += 32) {
    const v16h a = load2x8(xr + k0 + 8 * g, xr + k0 + 16 + 8 * g);
#pragma unroll
    for (int t = 0; t < 4; ++t) {
      const _Float16* wr = wqkvT + (size_t)(n0 + 16 * t + ln) * CCH + k0;
      v16h b = load2x8(wr + 8 * g, wr + 16 + 8 * g);
      acc[t] = wmma_f16(a, b, acc[t]);
    }
  }

  const float qscale = 0.17677669529663689f;  // 32^-0.5
#pragma unroll
  for (int t = 0; t < 4; ++t) {
    const int f    = n0 + 16 * t + ln;  // 0..383
    const int sel  = f >> 7;            // 0=q, 1=k, 2=v (uniform per block)
    const int ch   = f & 127;
    const int head = ch >> 5;
    const int dc   = ch & 31;
#pragma unroll
    for (int r = 0; r < 8; ++r) {            // C/D: M = r + 8g, N = ln
      const int m  = m0 + r + 8 * g;
      const int bb = m >> 12;                // / NTOK
      const int n  = m & (NTOK - 1);
      const size_t bh = (size_t)(bb * HEADS + head);
      if (sel == 0)
        q[(bh * NTOK + n) * DHEAD + dc] = (_Float16)(acc[t][r] * qscale);
      else if (sel == 1)
        kmat[(bh * NTOK + n) * DHEAD + dc] = (_Float16)acc[t][r];
      else
        vT[(bh * DHEAD + dc) * NTOK + n] = (_Float16)acc[t][r];
    }
  }
}

// ---------------------------------------------------------------------------
// Stage 2: flash attention. 256 threads = 8 waves; wave w owns Q rows
// [i0, i0+16). Per 64-token step the block stages K (64x32) and V^T (32x64)
// into double-buffered LDS (16B per thread per tile; async global->LDS when
// available), one workgroup barrier per iteration, then every wave runs
// 4 score WMMAs, online softmax, and 4 P.V WMMAs from LDS fragments.
// K/V global traffic drops 8x vs per-wave streaming.
// ---------------------------------------------------------------------------
__global__ void __launch_bounds__(256)
flash_attn_kernel(const _Float16* __restrict__ q,
                  const _Float16* __restrict__ kmat,
                  const _Float16* __restrict__ vT,
                  _Float16* __restrict__ att) {
  __shared__ __align__(16) _Float16 k_lds[2][JTILE * DHEAD];  // [tok][d], 2x4KB
  __shared__ __align__(16) _Float16 v_lds[2][DHEAD * JTILE];  // [d][tok], 2x4KB
  __shared__ __align__(16) _Float16 p_lds[8][16 * JTILE];     // per-wave P, 8x2KB

  const int tid  = threadIdx.x;
  const int wave = tid >> 5;
  const int lane = tid & 31;
  const int g  = lane >> 4;
  const int ln = lane & 15;
  const int bh = blockIdx.x;                          // b*HEADS + head
  const int i0 = blockIdx.y * QROWS_BLK + wave * 16;  // this wave's Q rows
  const int bb   = bh >> 2;
  const int head = bh & 3;

  const _Float16* qb = q    + (size_t)bh * NTOK * DHEAD;
  const _Float16* kb = kmat + (size_t)bh * NTOK * DHEAD;
  const _Float16* vb = vT   + (size_t)bh * DHEAD * NTOK;

  // Q A-fragment (held for the whole pass)
  const _Float16* qr = qb + (size_t)(i0 + ln) * DHEAD;
  const v16h aq = load2x8(qr + 8 * g, qr + 16 + 8 * g);

  // Cooperative staging coordinates for this thread (one 16B chunk per tile).
  const int kc_tok = tid >> 2;            // K: token row, 4 chunks/row
  const int kc_off = (tid & 3) * 8;       //    halfs offset within row
  const int vc_d   = tid >> 3;            // V: d row, 8 chunks/row
  const int vc_off = (tid & 7) * 8;

  float mrow[8], lsum[8];
#pragma unroll
  for (int r = 0; r < 8; ++r) { mrow[r] = -3.0e38f; lsum[r] = 0.f; }
  v8f o0 = {}, o1 = {};
  const v8f zero = {};
  const float LOG2E = 1.4426950408889634f;

  int buf = 0;
  for (int j0 = 0; j0 < NTOK; j0 += JTILE, buf ^= 1) {
    // ---- Stage K/V tile into LDS (all 256 threads, 16B each per tile) ----
    const _Float16* ksrc = kb + (size_t)(j0 + kc_tok) * DHEAD + kc_off;
    const _Float16* vsrc = vb + (size_t)vc_d * NTOK + j0 + vc_off;
#if USE_ASYNC_LDS
    // ASYNCcnt-tracked DMA directly into LDS; no VGPR round-trip.
    __builtin_amdgcn_global_load_async_to_lds_b128(
        (gv4i*)ksrc, (lv4i*)&k_lds[buf][tid * 8], 0, 0);
    __builtin_amdgcn_global_load_async_to_lds_b128(
        (gv4i*)vsrc, (lv4i*)&v_lds[buf][tid * 8], 0, 0);
#else
    *reinterpret_cast<v8h*>(&k_lds[buf][tid * 8]) =
        *reinterpret_cast<const v8h*>(ksrc);
    *reinterpret_cast<v8h*>(&v_lds[buf][tid * 8]) =
        *reinterpret_cast<const v8h*>(vsrc);
#endif
    if (j0 + JTILE < NTOK) {  // pull next tiles toward the WGP (global_prefetch_b8)
      __builtin_prefetch(kb + (size_t)(j0 + JTILE + kc_tok) * DHEAD + kc_off, 0, 0);
      __builtin_prefetch(vb + (size_t)vc_d * NTOK + j0 + JTILE + vc_off, 0, 0);
    }
#if USE_ASYNC_LDS
    __builtin_amdgcn_s_wait_asynccnt(0);  // this wave's tiles landed in LDS
#endif
    __syncthreads();  // all waves' tiles visible; also guards buffer re-use

    // ---- Scores: 4 column tiles of K^T from LDS ----
    v8f s[4];
#pragma unroll
    for (int t = 0; t < 4; ++t) {
      const _Float16* kr = &k_lds[buf][(16 * t + ln) * DHEAD];
      v16h bk = load2x8(kr + 8 * g, kr + 16 + 8 * g);
      s[t] = wmma_f16(aq, bk, zero);
    }

    // ---- Online softmax; row M = r + 8g spans the 16 lanes of this group ----
#pragma unroll
    for (int r = 0; r < 8; ++r) {
      float mx = fmaxf(fmaxf(s[0][r], s[1][r]), fmaxf(s[2][r], s[3][r]));
      mx = fmaxf(mx, __shfl_xor(mx, 1, 32));
      mx = fmaxf(mx, __shfl_xor(mx, 2, 32));
      mx = fmaxf(mx, __shfl_xor(mx, 4, 32));
      mx = fmaxf(mx, __shfl_xor(mx, 8, 32));
      const float mnew = fmaxf(mrow[r], mx);
      const float corr = exp2f((mrow[r] - mnew) * LOG2E);
      float p[4], ps = 0.f;
#pragma unroll
      for (int t = 0; t < 4; ++t) {
        p[t] = exp2f((s[t][r] - mnew) * LOG2E);
        ps += p[t];
      }
      ps += __shfl_xor(ps, 1, 32);
      ps += __shfl_xor(ps, 2, 32);
      ps += __shfl_xor(ps, 4, 32);
      ps += __shfl_xor(ps, 8, 32);
      lsum[r] = lsum[r] * corr + ps;
      mrow[r] = mnew;
      o0[r] *= corr;
      o1[r] *= corr;
      // stash P in this wave's row-major LDS tile (C-frag -> [M][col])
#pragma unroll
      for (int t = 0; t < 4; ++t)
        p_lds[wave][(r + 8 * g) * JTILE + 16 * t + ln] = (_Float16)p[t];
    }

    // ---- Re-read P as A fragments (per-wave region; DS in-order) ----
    const _Float16* pr = &p_lds[wave][ln * JTILE];
    v16h ap0 = load2x8(pr + 8 * g, pr + 16 + 8 * g);       // tokens j0+0..31
    v16h ap1 = load2x8(pr + 32 + 8 * g, pr + 48 + 8 * g);  // tokens j0+32..63

    // ---- O accumulation from the shared V tile ----
#pragma unroll
    for (int kblk = 0; kblk < 2; ++kblk) {
      const _Float16* vr0 = &v_lds[buf][ln * JTILE + 32 * kblk];        // d 0..15
      const _Float16* vr1 = &v_lds[buf][(16 + ln) * JTILE + 32 * kblk]; // d 16..31
      v16h bv0 = load2x8(vr0 + 8 * g, vr0 + 16 + 8 * g);
      v16h bv1 = load2x8(vr1 + 8 * g, vr1 + 16 + 8 * g);
      const v16h ap = kblk ? ap1 : ap0;
      o0 = wmma_f16(ap, bv0, o0);
      o1 = wmma_f16(ap, bv1, o1);
    }
  }

  // ---- Normalize and store to att[b][token][head*32 + d] as f16 ----
#pragma unroll
  for (int r = 0; r < 8; ++r) {
    const float inv = 1.0f / lsum[r];
    const int m = i0 + r + 8 * g;
    _Float16* dst = att + ((size_t)bb * NTOK + m) * (HEADS * DHEAD) + head * DHEAD;
    dst[ln]      = (_Float16)(o0[r] * inv);
    dst[16 + ln] = (_Float16)(o1[r] * inv);
  }
}

// ---------------------------------------------------------------------------
// Stage 3: out = att @ w_out + b_out, 16(M) x 64(N) tile per wave, f32 output.
// woutT is [128][128] f16 column-major.
// ---------------------------------------------------------------------------
__global__ void out_proj_kernel(const _Float16* __restrict__ att,
                                const _Float16* __restrict__ woutT,
                                const float* __restrict__ bout,
                                float* __restrict__ out) {
  const int lane = threadIdx.x & 31;
  const int g  = lane >> 4;
  const int ln = lane & 15;
  const int m0 = blockIdx.x * 16;
  const int n0 = blockIdx.y * 64;

  v8f acc[4] = {};
  const _Float16* ar = att + (size_t)(m0 + ln) * CCH;
#pragma unroll
  for (int k0 = 0; k0 < CCH; k0 += 32) {
    const v16h a = load2x8(ar + k0 + 8 * g, ar + k0 + 16 + 8 * g);
#pragma unroll
    for (int t = 0; t < 4; ++t) {
      const _Float16* wr = woutT + (size_t)(n0 + 16 * t + ln) * CCH + k0;
      v16h b = load2x8(wr + 8 * g, wr + 16 + 8 * g);
      acc[t] = wmma_f16(a, b, acc[t]);
    }
  }
#pragma unroll
  for (int t = 0; t < 4; ++t) {
    const float bias = bout[n0 + 16 * t + ln];
#pragma unroll
    for (int r = 0; r < 8; ++r)
      out[(size_t)(m0 + r + 8 * g) * CCH + n0 + 16 * t + ln] = acc[t][r] + bias;
  }
}

// ---------------------------------------------------------------------------
// Host launch. Workspace layout (f16 elements, ~20.2 MB total):
//   [0)        q      : BATCH*HEADS*NTOK*DHEAD = 2 Mi halfs
//   [2 Mi)     k      : 2 Mi halfs
//   [4 Mi)     vT     : 2 Mi halfs ([b,H,d,N])
//   [6 Mi)     att    : MROWS*CCH = 2 Mi halfs
//   [8 Mi)     xh     : MROWS*CCH = 2 Mi halfs (f16 copy of x)
//   [10 Mi)    wqkvT  : 384*128 halfs (col-major f16 w_qkv)
//   [+49152)   woutT  : 128*128 halfs (col-major f16 w_out)
// ---------------------------------------------------------------------------
extern "C" void kernel_launch(void* const* d_in, const int* in_sizes, int n_in,
                              void* d_out, int out_size, void* d_ws, size_t ws_size,
                              hipStream_t stream) {
  (void)in_sizes; (void)n_in; (void)out_size; (void)ws_size;
  const float* x    = (const float*)d_in[0];
  const float* wqkv = (const float*)d_in[1];
  const float* wout = (const float*)d_in[2];
  const float* bout = (const float*)d_in[3];
  float* out = (float*)d_out;

  const size_t QKV_ELEMS = (size_t)BATCH * HEADS * NTOK * DHEAD;  // 2 Mi
  _Float16* ws    = (_Float16*)d_ws;
  _Float16* q     = ws;
  _Float16* km    = ws + QKV_ELEMS;
  _Float16* vT    = ws + 2 * QKV_ELEMS;
  _Float16* att   = ws + 3 * QKV_ELEMS;
  _Float16* xh    = ws + 4 * QKV_ELEMS;
  _Float16* wqkvT = ws + 5 * QKV_ELEMS;
  _Float16* woutT = wqkvT + (size_t)F_QKV * CCH;

  // Stage 0: one-time precision/layout conversion (memory-bound, tiny).
  const int NX = MROWS * CCH;  // 2 Mi elements
  cvt_f16_kernel<<<dim3((NX + 255) / 256), 256, 0, stream>>>(x, xh, NX);
  tcvt_f16_kernel<<<dim3((CCH * F_QKV + 255) / 256), 256, 0, stream>>>(wqkv, wqkvT, CCH, F_QKV);
  tcvt_f16_kernel<<<dim3((CCH * CCH + 255) / 256), 256, 0, stream>>>(wout, woutT, CCH, CCH);

  // Stage 1-3: all matmuls on the WMMA pipes.
  qkv_proj_kernel<<<dim3(MROWS / 16, F_QKV / 64), 32, 0, stream>>>(xh, wqkvT, q, km, vT);
  flash_attn_kernel<<<dim3(BATCH * HEADS, NTOK / QROWS_BLK), 256, 0, stream>>>(q, km, vT, att);
  out_proj_kernel<<<dim3(MROWS / 16, CCH / 64), 32, 0, stream>>>(att, woutT, bout, out);
}
